// RGCNModel_32100585570900
// MI455X (gfx1250) — matrix operand order repaired and checked by reading, not compiled, not run on hardware
//
#include <hip/hip_runtime.h>

#define RR 8
#define FI 128                 // feature dim per relation block
#define KTOT ((RR + 1) * FI)   // 1152 fused-K
#define NT 4                   // col tiles per wave -> 64 output cols per block
#define LBS 136                // LDS row stride (ushorts): 68 dwords -> conflict-free

typedef __attribute__((ext_vector_type(16))) __bf16 v16bf;
typedef __attribute__((ext_vector_type(8)))  float  v8f;

__device__ __forceinline__ unsigned short f2bf(float f) {  // RNE, for host-side-ish packing
  unsigned int u = __float_as_uint(f);
  return (unsigned short)((u + 0x7FFFu + ((u >> 16) & 1u)) >> 16);
}

// ---- per-(relation,dst) edge counts ----
__global__ void k_count(const int* __restrict__ et, const int* __restrict__ dst,
                        float* __restrict__ cnt, int E, int N) {
  int e = blockIdx.x * blockDim.x + threadIdx.x;
  if (e < E) atomicAdd(&cnt[(size_t)et[e] * N + dst[e]], 1.0f);
}

__global__ void k_invert(float* __restrict__ c, int n) {  // c <- 1/max(c,1)
  int i = blockIdx.x * blockDim.x + threadIdx.x;
  if (i < n) c[i] = 1.0f / fmaxf(c[i], 1.0f);
}

// ---- pack fused weights transposed to bf16: Bt[o][r*128+i]=W[r][i][o]; Bt[o][1024+i]=root[i][o]
__global__ void k_packw(const float* __restrict__ W, const float* __restrict__ root,
                        unsigned short* __restrict__ Bt, int O) {
  int total = O * KTOT;
  for (int idx = blockIdx.x * blockDim.x + threadIdx.x; idx < total;
       idx += gridDim.x * blockDim.x) {
    int o = idx / KTOT, k = idx - o * KTOT;
    int r = k >> 7, i = k & 127;
    float v = (r < RR) ? W[((size_t)r * FI + i) * O + o] : root[(size_t)i * O + o];
    Bt[(size_t)o * KTOT + k] = f2bf(v);
  }
}

// ---- edge scatter: one wave per edge, lane covers 4 features ----
__global__ void k_scatter(const int* __restrict__ src, const int* __restrict__ dst,
                          const int* __restrict__ et, const float* __restrict__ hin,
                          float* __restrict__ s, int E, int N) {
  int e = blockIdx.x * 8 + (threadIdx.x >> 5);
  if (e >= E) return;
  int lane = threadIdx.x & 31;
  int sn = src[e], r = et[e], dn = dst[e];
  const float4 v = ((const float4*)(hin + (size_t)sn * FI))[lane];
  float* sp = s + ((size_t)r * N + dn) * FI + lane * 4;
  atomicAdd(sp + 0, v.x); atomicAdd(sp + 1, v.y);
  atomicAdd(sp + 2, v.z); atomicAdd(sp + 3, v.w);
}

__device__ __forceinline__ v16bf cvt_a(const float4& f0, const float4& f1,
                                       const float4& f2, const float4& f3, float sc) {
  v16bf a;
  a[0]  = (__bf16)(f0.x * sc); a[1]  = (__bf16)(f0.y * sc);
  a[2]  = (__bf16)(f0.z * sc); a[3]  = (__bf16)(f0.w * sc);
  a[4]  = (__bf16)(f1.x * sc); a[5]  = (__bf16)(f1.y * sc);
  a[6]  = (__bf16)(f1.z * sc); a[7]  = (__bf16)(f1.w * sc);
  a[8]  = (__bf16)(f2.x * sc); a[9]  = (__bf16)(f2.y * sc);
  a[10] = (__bf16)(f2.z * sc); a[11] = (__bf16)(f2.w * sc);
  a[12] = (__bf16)(f3.x * sc); a[13] = (__bf16)(f3.y * sc);
  a[14] = (__bf16)(f3.z * sc); a[15] = (__bf16)(f3.w * sc);
  return a;
}

// ---- fused mean + GEMM + bias (+ReLU) via v_wmma_f32_16x16x32_bf16 ----
// Block: 256 thr = 8 waves; wave = 16 rows x 64 cols; outer loop over 9 K-blocks of 128.
__global__ __launch_bounds__(256) void
k_gemm(const float* __restrict__ s, const float* __restrict__ inv,
       const float* __restrict__ hin, const unsigned short* __restrict__ Bt,
       const float* __restrict__ bias, float* __restrict__ out,
       int N, int O, int relu) {
  __shared__ unsigned short lb[NT * 16 * LBS];  // 64 cols x 128 K panel, padded
  const int tid = threadIdx.x;
  const int wave = tid >> 5, lane = tid & 31;
  const int lanelo = lane & 15, half = lane >> 4;
  const int rowBase = blockIdx.x * 128 + wave * 16;
  const int colBase = blockIdx.y * (NT * 16);
  const int rowC = min(rowBase + lanelo, N - 1);  // clamped for loads

  v8f acc[NT];
#pragma unroll
  for (int t = 0; t < NT; t++) acc[t] = (v8f){0, 0, 0, 0, 0, 0, 0, 0};

  // cooperative B staging: thread t copies 64B; col = t>>2, seg = t&3
  const int stCol = tid >> 2, stSeg = tid & 3;
  const uint4* bsrcBase =
      (const uint4*)(Bt + (size_t)(colBase + stCol) * KTOT + stSeg * 32);
  uint4* bdst = (uint4*)(lb + stCol * LBS + stSeg * 32);

  const float* ap  = s + (size_t)rowC * FI;  // advances by N*FI per relation
  const float* ivp = inv + rowC;             // advances by N per relation

  for (int r = 0; r < RR + 1; ++r) {
    __syncthreads();
    {  // stage 64x128 bf16 B panel for this K-block (16KB, read-once per block)
      const uint4* bsrc = (const uint4*)((const unsigned short*)bsrcBase + r * FI);
      bdst[0] = bsrc[0]; bdst[1] = bsrc[1]; bdst[2] = bsrc[2]; bdst[3] = bsrc[3];
    }
    __syncthreads();

    const float* arow; float sc;
    if (r < RR) { arow = ap; sc = *ivp; }          // fold mean into A
    else        { arow = hin + (size_t)rowC * FI; sc = 1.0f; }  // root block

#pragma unroll
    for (int s4 = 0; s4 < 4; ++s4) {  // 4 x K=32 steps inside the 128 block
      // A fragment: lane row = lane&15; K chunks [b0..b0+7],[b0+16..b0+23], b0=half*8
      int ib = s4 * 32 + half * 8;
      const float4 f0 = *(const float4*)(arow + ib);
      const float4 f1 = *(const float4*)(arow + ib + 4);
      const float4 f2 = *(const float4*)(arow + ib + 16);
      const float4 f3 = *(const float4*)(arow + ib + 20);
      v16bf a = cvt_a(f0, f1, f2, f3, sc);

      // preload ALL NT B fragments, then issue WMMAs back-to-back
      union { v16bf v; uint4 q[2]; } bu[NT];
#pragma unroll
      for (int t = 0; t < NT; t++) {
        const unsigned short* bp = &lb[(t * 16 + lanelo) * LBS + s4 * 32 + half * 8];
        bu[t].q[0] = *(const uint4*)bp;
        bu[t].q[1] = *(const uint4*)(bp + 16);
      }
#pragma unroll
      for (int t = 0; t < NT; t++)
        acc[t] = __builtin_amdgcn_wmma_f32_16x16x32_bf16(
            false, a, false, bu[t].v, (short)0, acc[t], false, false);
    }
    ap += (size_t)N * FI;
    ivp += N;
  }

  // D layout: VGPR v -> row (lane>>4)*8+v, col lane&15; fuse bias+ReLU
#pragma unroll
  for (int t = 0; t < NT; t++) {
    int col = colBase + t * 16 + lanelo;
    float bv = bias[col];
#pragma unroll
    for (int v = 0; v < 8; v++) {
      int row = rowBase + half * 8 + v;
      if (row < N) {
        float val = acc[t][v] + bv;
        if (relu) val = fmaxf(val, 0.0f);
        out[(size_t)row * O + col] = val;
      }
    }
  }
}

// ---- log_softmax over 64 classes, one wave per node ----
__global__ void k_logsoftmax(const float* __restrict__ logits,
                             float* __restrict__ out, int N) {
  int node = blockIdx.x * 8 + (threadIdx.x >> 5);
  if (node >= N) return;
  int lane = threadIdx.x & 31;
  const float* p = logits + (size_t)node * 64;
  float a = p[lane], b = p[lane + 32];
  float m = fmaxf(a, b);
#pragma unroll
  for (int i = 16; i > 0; i >>= 1) m = fmaxf(m, __shfl_xor(m, i));
  float sum = __expf(a - m) + __expf(b - m);
#pragma unroll
  for (int i = 16; i > 0; i >>= 1) sum += __shfl_xor(sum, i);
  float lse = m + __logf(sum);
  out[(size_t)node * 64 + lane] = a - lse;
  out[(size_t)node * 64 + lane + 32] = b - lse;
}

extern "C" void kernel_launch(void* const* d_in, const int* in_sizes, int n_in,
                              void* d_out, int out_size, void* d_ws, size_t ws_size,
                              hipStream_t stream) {
  const float* x    = (const float*)d_in[0];
  const int*   eidx = (const int*)d_in[1];
  const int*   et   = (const int*)d_in[2];
  const float* W[4]  = {(const float*)d_in[3], (const float*)d_in[6],
                        (const float*)d_in[9], (const float*)d_in[12]};
  const float* rt[4] = {(const float*)d_in[4], (const float*)d_in[7],
                        (const float*)d_in[10], (const float*)d_in[13]};
  const float* bs[4] = {(const float*)d_in[5], (const float*)d_in[8],
                        (const float*)d_in[11], (const float*)d_in[14]};
  const int N = in_sizes[0] / FI;
  const int E = in_sizes[2];
  const int* srcp = eidx;        // edge_index[0,:]
  const int* dstp = eidx + E;    // edge_index[1,:]

  char* ws = (char*)d_ws;
  size_t off = 0;
  auto alloc = [&](size_t bytes) -> char* {
    char* p = ws + off; off += (bytes + 255) & ~(size_t)255; return p;
  };
  float* cnt  = (float*)alloc((size_t)RR * N * sizeof(float));       // -> inverse counts
  float* sbuf = (float*)alloc((size_t)RR * N * FI * sizeof(float));  // f32 aggregation
  float* hA   = (float*)alloc((size_t)N * FI * sizeof(float));
  float* hB   = (float*)alloc((size_t)N * FI * sizeof(float));
  int Odim[4] = {128, 128, 128, 64};
  unsigned short* Bt[4];
  for (int l = 0; l < 4; l++)
    Bt[l] = (unsigned short*)alloc((size_t)Odim[l] * KTOT * sizeof(unsigned short));

  // counts -> 1/max(c,1), computed once (graph topology fixed across layers)
  hipMemsetAsync(cnt, 0, (size_t)RR * N * sizeof(float), stream);
  k_count<<<(E + 255) / 256, 256, 0, stream>>>(et, dstp, cnt, E, N);
  k_invert<<<((RR * N) + 255) / 256, 256, 0, stream>>>(cnt, RR * N);

  for (int l = 0; l < 4; l++)
    k_packw<<<512, 256, 0, stream>>>(W[l], rt[l], Bt[l], Odim[l]);

  const float* hin = x;
  float* houts[4] = {hA, hB, hA, hB};  // layer4 (O=64) fits in hB
  for (int l = 0; l < 4; l++) {
    hipMemsetAsync(sbuf, 0, (size_t)RR * N * FI * sizeof(float), stream);
    k_scatter<<<(E + 7) / 8, 256, 0, stream>>>(srcp, dstp, et, hin, sbuf, E, N);
    int O = Odim[l];
    dim3 grid((N + 127) / 128, O / (NT * 16));
    k_gemm<<<grid, 256, 0, stream>>>(sbuf, cnt, hin, Bt[l], bs[l], houts[l],
                                     N, O, l < 3 ? 1 : 0);
    hin = houts[l];
  }
  k_logsoftmax<<<(N + 7) / 8, 256, 0, stream>>>(houts[3], (float*)d_out, N);
}
